// DITAttention_63797444215713
// MI455X (gfx1250) — matrix operand order repaired and checked by reading, not compiled, not run on hardware
//
#include <hip/hip_runtime.h>
#include <hip/hip_bf16.h>

typedef __attribute__((ext_vector_type(16))) __bf16 v16bf;
typedef __attribute__((ext_vector_type(8)))  __bf16 v8bf;
typedef __attribute__((ext_vector_type(8)))  float  v8f;

#define HIDDEN 1024
#define NHEADS 16
#define HDIM   64
#define SEQ    2048
#define BATCH  2
#define TOKENS (BATCH * SEQ)

// gfx1250 async global->LDS staging (ASYNCcnt path). Flip to 0 for sync fallback.
#define USE_ASYNC_LDS 1

// ---------- helpers ----------

static __device__ __forceinline__ unsigned short f2bf(float f) {
  unsigned u = __float_as_uint(f);
  u += 0x7FFFu + ((u >> 16) & 1u);   // round-to-nearest-even
  return (unsigned short)(u >> 16);
}

// Copy one 16-byte chunk global->LDS. Async variant uses
// GLOBAL_LOAD_ASYNC_TO_LDS_B128 (tracked by ASYNCcnt); dsaddr = low 32 bits of
// the generic LDS pointer (LDS aperture offset).
static __device__ __forceinline__ void copy16_g2l(const unsigned short* __restrict__ src,
                                                  unsigned short* ldst) {
#if USE_ASYNC_LDS
  const unsigned lds_off = (unsigned)(size_t)ldst;
  asm volatile("global_load_async_to_lds_b128 %0, %1, off"
               :: "v"(lds_off), "v"(src) : "memory");
#else
  *(v8bf*)ldst = *(const v8bf*)src;
#endif
}

static __device__ __forceinline__ void wait_async() {
#if USE_ASYNC_LDS
  asm volatile("s_wait_asynccnt 0" ::: "memory");
#endif
}

// A-operand (16x32 MxK bf16) from an LDS tile with row stride 32 halves.
// lanes 0-15: K = {0..7, 16..23}; lanes 16-31: {8..15, 24..31}
static __device__ __forceinline__ v16bf fragA_lds(const unsigned short* tile /*[rows][32]*/,
                                                  int row0, int lane) {
  const int r    = row0 + (lane & 15);
  const int koff = (lane & 16) ? 8 : 0;
  const unsigned short* q = tile + r * 32 + koff;
  v8bf lo = *(const v8bf*)q;
  v8bf hi = *(const v8bf*)(q + 16);
  v16bf a;
#pragma unroll
  for (int i = 0; i < 8; ++i) { a[i] = lo[i]; a[i + 8] = hi[i]; }
  return a;
}

// B-operand (32x16 KxN bf16) from an LDS tile, row stride `ld` halves:
// lane -> column N (tile row n0+lane&15); lanes 0-15 K={0..7,16..23}, lanes 16-31 {8..15,24..31}
static __device__ __forceinline__ v16bf fragB_lds(const unsigned short* tile,
                                                  int ld, int n0, int k0, int lane) {
  const int n    = n0 + (lane & 15);
  const int koff = (lane & 16) ? 8 : 0;
  const unsigned short* q = tile + n * ld + k0 + koff;
  v8bf lo = *(const v8bf*)q;
  v8bf hi = *(const v8bf*)(q + 16);
  v16bf b;
#pragma unroll
  for (int i = 0; i < 8; ++i) { b[i] = lo[i]; b[i + 8] = hi[i]; }
  return b;
}

// A-operand directly from global (row stride ld halves)
static __device__ __forceinline__ v16bf fragA_glb(const unsigned short* __restrict__ p,
                                                  int ld, int row0, int k0, int lane) {
  const int r    = row0 + (lane & 15);
  const int koff = (lane & 16) ? 8 : 0;
  const unsigned short* q = p + (size_t)r * ld + k0 + koff;
  v8bf lo = *(const v8bf*)q;
  v8bf hi = *(const v8bf*)(q + 16);
  v16bf a;
#pragma unroll
  for (int i = 0; i < 8; ++i) { a[i] = lo[i]; a[i + 8] = hi[i]; }
  return a;
}

static __device__ __forceinline__ v8f wmma_bf16(v16bf a, v16bf b, v8f c) {
  return __builtin_amdgcn_wmma_f32_16x16x32_bf16(false, a, false, b, (short)0, c,
                                                 false, false);
}

// ---------- stage 0: f32 -> bf16 ----------

__global__ void cvt_f32_to_bf16(const float* __restrict__ src,
                                unsigned short* __restrict__ dst, int n) {
  int i = blockIdx.x * blockDim.x + threadIdx.x;
  int stride = gridDim.x * blockDim.x;
  for (; i < n; i += stride) {
    if (i + stride < n) __builtin_prefetch(src + i + stride, 0, 3);
    dst[i] = f2bf(src[i]);
  }
}

// ---------- stage 1: fused QKV projection + RoPE ----------
// grid: (TOKENS/64, NHEADS, 3[q,k,v]), block 128 (4 waves).
// Block computes a 64x64 tile; A(64x32) and B(64x32) k-step tiles are
// double-buffered in LDS via async copies. Q,K -> [b,h,s,d] (RoPE'd);
// V -> transposed [b,h,d,s].

__global__ __launch_bounds__(128) void qkv_rope_kernel(
    const unsigned short* __restrict__ Xb,
    const unsigned short* __restrict__ Wqb,
    const unsigned short* __restrict__ Wkb,
    const unsigned short* __restrict__ Wvb,
    unsigned short* __restrict__ Qb,
    unsigned short* __restrict__ Kb,
    unsigned short* __restrict__ Vtb) {
  const int tid  = threadIdx.x;
  const int lane = tid & 31;
  const int wave = tid >> 5;
  const int mblk = blockIdx.x * 64;
  const int h    = blockIdx.y;
  const int z    = blockIdx.z;
  const unsigned short* W = (z == 0) ? Wqb : (z == 1) ? Wkb : Wvb;
  const int n0 = h * HDIM;

  __shared__ __align__(16) unsigned short tA[2][64][32];
  __shared__ __align__(16) unsigned short tB[2][64][32];

  // 64x32 tiles = 256 chunks of 16B; 128 threads -> 2 unconditional chunks each
  auto stage = [&](int nb, int k0) {
#pragma unroll
    for (int i = 0; i < 2; ++i) {
      const int c = tid + i * 128;
      const int row = c >> 2, cc = (c & 3) * 8;
      copy16_g2l(Xb + (size_t)(mblk + row) * HIDDEN + k0 + cc, &tA[nb][row][cc]);
      copy16_g2l(W  + (size_t)(n0 + row)  * HIDDEN + k0 + cc, &tB[nb][row][cc]);
    }
  };

  stage(0, 0);

  v8f acc[4] = {};
  for (int k0 = 0; k0 < HIDDEN; k0 += 32) {
    const int buf = (k0 >> 5) & 1;
    wait_async();
    __syncthreads();
    if (k0 + 32 < HIDDEN) stage(buf ^ 1, k0 + 32);

    v16bf a = fragA_lds(&tA[buf][0][0], wave * 16, lane);
#pragma unroll
    for (int nt = 0; nt < 4; ++nt) {
      v16bf bfr = fragB_lds(&tB[buf][0][0], 32, nt * 16, 0, lane);
      acc[nt] = wmma_bf16(a, bfr, acc[nt]);
    }
    __syncthreads();
  }

  // C layout: lane holds column (n = lane&15); rows j + 8*(lane>=16)
  const int mrow  = mblk + wave * 16;
  const int half  = lane >> 4;
  const int col   = lane & 15;
  const int bidx  = mrow >> 11;          // batch
  const int sbase = mrow & (SEQ - 1);    // seq position base

  if (z == 2) {
    unsigned short* Vt = Vtb + (size_t)(bidx * NHEADS + h) * HDIM * SEQ;
#pragma unroll
    for (int dt = 0; dt < 4; ++dt) {
      const int d = dt * 16 + col;
#pragma unroll
      for (int r = 0; r < 8; ++r) {
        const int s = sbase + r + 8 * half;
        Vt[(size_t)d * SEQ + s] = f2bf(acc[dt][r]);
      }
    }
  } else {
    unsigned short* dst = ((z == 0) ? Qb : Kb) + (size_t)(bidx * NHEADS + h) * SEQ * HDIM;
#pragma unroll
    for (int dt = 0; dt < 4; ++dt) {
      const int d = dt * 16 + col;
      // inv_freq = 10000^(-(d&~1)/64) = exp2(-(d>>1) * log2(10000)/32)
      const float invf = __builtin_amdgcn_exp2f(-(float)(d >> 1) * 0.4152410118609253f);
#pragma unroll
      for (int r = 0; r < 8; ++r) {
        const int s = sbase + r + 8 * half;
        const float ang = (float)s * invf;
        const float cs = __cosf(ang);
        const float sn = __sinf(ang);
        const float x  = acc[dt][r];
        const float po = __shfl_xor(x, 1, 32);
        const float v = (lane & 1) ? fmaf(po, sn, x * cs) : fmaf(-po, sn, x * cs);
        dst[(size_t)s * HDIM + d] = f2bf(v);
      }
    }
  }
}

// ---------- stage 2: flash attention ----------
// grid: (SEQ/64, NHEADS, BATCH), block 128 (4 waves), one 16-query tile/wave.
// Per 32-key block: K (32x64) and V (64x32) tiles shared by all 4 waves are
// async-staged to LDS, double-buffered. Online (base-2) softmax.

__global__ __launch_bounds__(128) void flash_attn_kernel(
    const unsigned short* __restrict__ Qb,
    const unsigned short* __restrict__ Kb,
    const unsigned short* __restrict__ Vtb,
    unsigned short* __restrict__ AOb) {
  const int tid  = threadIdx.x;
  const int lane = tid & 31;
  const int wave = tid >> 5;
  const int q0   = blockIdx.x * 64 + wave * 16;
  const int h    = blockIdx.y;
  const int b    = blockIdx.z;

  const unsigned short* Qh = Qb  + (size_t)(b * NHEADS + h) * SEQ * HDIM;
  const unsigned short* Kh = Kb  + (size_t)(b * NHEADS + h) * SEQ * HDIM;
  const unsigned short* Vh = Vtb + (size_t)(b * NHEADS + h) * HDIM * SEQ;

  __shared__ __align__(16) unsigned short tK[2][32][64];   // [t_local][d]
  __shared__ __align__(16) unsigned short tV[2][64][32];   // [d][t_local]
  __shared__ __align__(16) unsigned short Pld[4][16][32];  // per-wave P staging

  auto stage = [&](int nb, int t0) {
    // K tile: 32 rows x 64 halves (8 chunks/row); V tile: 64 rows x 32 halves
#pragma unroll
    for (int i = 0; i < 2; ++i) {
      const int c = tid + i * 128;
      const int kr = c >> 3, kc = (c & 7) * 8;
      copy16_g2l(Kh + (size_t)(t0 + kr) * HDIM + kc, &tK[nb][kr][kc]);
      const int vr = c >> 2, vc = (c & 3) * 8;
      copy16_g2l(Vh + (size_t)vr * SEQ + t0 + vc, &tV[nb][vr][vc]);
    }
  };

  const v16bf qf0 = fragA_glb(Qh, HDIM, q0, 0, lane);
  const v16bf qf1 = fragA_glb(Qh, HDIM, q0, 32, lane);

  stage(0, 0);

  float m[8], l[8];
#pragma unroll
  for (int r = 0; r < 8; ++r) { m[r] = -3.0e38f; l[r] = 0.0f; }
  v8f o[4] = {};

  const float sc  = 0.18033688011112042f;  // (1/sqrt(64)) * log2(e)
  const int half  = lane >> 4;
  const int col   = lane & 15;

  for (int t0 = 0; t0 < SEQ; t0 += 32) {
    const int buf = (t0 >> 5) & 1;
    wait_async();
    __syncthreads();
    if (t0 + 32 < SEQ) stage(buf ^ 1, t0 + 32);

    // ---- scores: S = Q * K^T (16x32 tile as two 16x16 accumulators)
    v8f s0 = {}, s1 = {};
    {
      v16bf k00 = fragB_lds(&tK[buf][0][0], 64, 0,  0,  lane);
      v16bf k01 = fragB_lds(&tK[buf][0][0], 64, 0,  32, lane);
      s0 = wmma_bf16(qf0, k00, s0);
      s0 = wmma_bf16(qf1, k01, s0);
      v16bf k10 = fragB_lds(&tK[buf][0][0], 64, 16, 0,  lane);
      v16bf k11 = fragB_lds(&tK[buf][0][0], 64, 16, 32, lane);
      s1 = wmma_bf16(qf0, k10, s1);
      s1 = wmma_bf16(qf1, k11, s1);
    }

    // ---- online softmax; rows reduce across the 16 lanes of each half
#pragma unroll
    for (int r = 0; r < 8; ++r) {
      float x0 = s0[r] * sc;
      float x1 = s1[r] * sc;
      float tm = fmaxf(x0, x1);
#pragma unroll
      for (int msk = 1; msk <= 8; msk <<= 1) tm = fmaxf(tm, __shfl_xor(tm, msk, 32));
      const float mn   = fmaxf(m[r], tm);
      const float corr = __builtin_amdgcn_exp2f(m[r] - mn);
      const float p0   = __builtin_amdgcn_exp2f(x0 - mn);
      const float p1   = __builtin_amdgcn_exp2f(x1 - mn);
      float ps = p0 + p1;
#pragma unroll
      for (int msk = 1; msk <= 8; msk <<= 1) ps += __shfl_xor(ps, msk, 32);
      l[r] = l[r] * corr + ps;
      m[r] = mn;
#pragma unroll
      for (int dt = 0; dt < 4; ++dt) o[dt][r] *= corr;
      Pld[wave][r + 8 * half][col]      = f2bf(p0);
      Pld[wave][r + 8 * half][16 + col] = f2bf(p1);
    }
    asm volatile("s_wait_dscnt 0" ::: "memory");

    // ---- reload P in A-operand layout (lane -> query row, split-K over t)
    v16bf pf;
    {
      const unsigned short* q = &Pld[wave][col][(lane & 16) ? 8 : 0];
      v8bf lo = *(const v8bf*)q;
      v8bf hi = *(const v8bf*)(q + 16);
#pragma unroll
      for (int i = 0; i < 8; ++i) { pf[i] = lo[i]; pf[i + 8] = hi[i]; }
    }

    // ---- O += P * V
#pragma unroll
    for (int dt = 0; dt < 4; ++dt) {
      v16bf vf = fragB_lds(&tV[buf][0][0], 32, dt * 16, 0, lane);
      o[dt] = wmma_bf16(pf, vf, o[dt]);
    }
    __syncthreads();
  }

  // ---- normalize and store per-head output into [b*S+s][h*64+d] (bf16)
  unsigned short* out = AOb + ((size_t)(b * SEQ) + q0) * HIDDEN + h * HDIM;
#pragma unroll
  for (int dt = 0; dt < 4; ++dt) {
    const int d = dt * 16 + col;
#pragma unroll
    for (int r = 0; r < 8; ++r) {
      const int srel = r + 8 * half;
      out[(size_t)srel * HIDDEN + d] = f2bf(o[dt][r] / l[r]);
    }
  }
}

// ---------- stage 3: output projection (f32 out) ----------

__global__ __launch_bounds__(128) void oproj_kernel(
    const unsigned short* __restrict__ AOb,
    const unsigned short* __restrict__ Wob,
    float* __restrict__ out) {
  const int tid  = threadIdx.x;
  const int lane = tid & 31;
  const int wave = tid >> 5;
  const int mblk = blockIdx.x * 64;
  const int n0   = blockIdx.y * 64;

  __shared__ __align__(16) unsigned short tA[2][64][32];
  __shared__ __align__(16) unsigned short tB[2][64][32];

  auto stage = [&](int nb, int k0) {
#pragma unroll
    for (int i = 0; i < 2; ++i) {
      const int c = tid + i * 128;
      const int row = c >> 2, cc = (c & 3) * 8;
      copy16_g2l(AOb + (size_t)(mblk + row) * HIDDEN + k0 + cc, &tA[nb][row][cc]);
      copy16_g2l(Wob + (size_t)(n0 + row)  * HIDDEN + k0 + cc, &tB[nb][row][cc]);
    }
  };

  stage(0, 0);

  v8f acc[4] = {};
  for (int k0 = 0; k0 < HIDDEN; k0 += 32) {
    const int buf = (k0 >> 5) & 1;
    wait_async();
    __syncthreads();
    if (k0 + 32 < HIDDEN) stage(buf ^ 1, k0 + 32);

    v16bf a = fragA_lds(&tA[buf][0][0], wave * 16, lane);
#pragma unroll
    for (int nt = 0; nt < 4; ++nt) {
      v16bf bfr = fragB_lds(&tB[buf][0][0], 32, nt * 16, 0, lane);
      acc[nt] = wmma_bf16(a, bfr, acc[nt]);
    }
    __syncthreads();
  }

  const int mrow = mblk + wave * 16;
  const int half = lane >> 4;
  const int col  = lane & 15;
#pragma unroll
  for (int dt = 0; dt < 4; ++dt) {
#pragma unroll
    for (int r = 0; r < 8; ++r) {
      out[(size_t)(mrow + r + 8 * half) * HIDDEN + n0 + dt * 16 + col] = acc[dt][r];
    }
  }
}

// ---------- launcher ----------

extern "C" void kernel_launch(void* const* d_in, const int* in_sizes, int n_in,
                              void* d_out, int out_size, void* d_ws, size_t ws_size,
                              hipStream_t stream) {
  const float* X  = (const float*)d_in[0];
  const float* Wq = (const float*)d_in[1];
  const float* Wk = (const float*)d_in[2];
  const float* Wv = (const float*)d_in[3];
  const float* Wo = (const float*)d_in[4];

  const size_t NX = (size_t)TOKENS * HIDDEN;     // 4M elems
  const size_t NW = (size_t)HIDDEN * HIDDEN;     // 1M elems

  unsigned short* p   = (unsigned short*)d_ws;
  unsigned short* Xb  = p; p += NX;
  unsigned short* Wqb = p; p += NW;
  unsigned short* Wkb = p; p += NW;
  unsigned short* Wvb = p; p += NW;
  unsigned short* Wob = p; p += NW;
  unsigned short* Qb  = p; p += NX;   // [b,h,s,d]
  unsigned short* Kb  = p; p += NX;   // [b,h,s,d]
  unsigned short* Vtb = p; p += NX;   // [b,h,d,s]
  unsigned short* AOb = p; p += NX;   // [b*s, hidden]

  cvt_f32_to_bf16<<<2048, 256, 0, stream>>>(X,  Xb,  (int)NX);
  cvt_f32_to_bf16<<<1024, 256, 0, stream>>>(Wq, Wqb, (int)NW);
  cvt_f32_to_bf16<<<1024, 256, 0, stream>>>(Wk, Wkb, (int)NW);
  cvt_f32_to_bf16<<<1024, 256, 0, stream>>>(Wv, Wvb, (int)NW);
  cvt_f32_to_bf16<<<1024, 256, 0, stream>>>(Wo, Wob, (int)NW);

  qkv_rope_kernel<<<dim3(TOKENS / 64, NHEADS, 3), 128, 0, stream>>>(
      Xb, Wqb, Wkb, Wvb, Qb, Kb, Vtb);

  flash_attn_kernel<<<dim3(SEQ / 64, NHEADS, BATCH), 128, 0, stream>>>(
      Qb, Kb, Vtb, AOb);

  oproj_kernel<<<dim3(TOKENS / 64, NHEADS), 128, 0, stream>>>(
      AOb, Wob, (float*)d_out);
}
